// GroupQueryAttention_678604832925
// MI455X (gfx1250) — matrix operand order repaired and checked by reading, not compiled, not run on hardware
//
#include <hip/hip_runtime.h>
#include <hip/hip_bf16.h>
#include <math.h>

// ---------------------------------------------------------------------------
// CDNA5 (gfx1250) GQA transformer block.
// bf16 weights staged in workspace (w_up fits 192MB L2 only as bf16), all
// matmuls via v_wmma_f32_16x16x32_bf16 (f32 accum), flash-style wave32
// attention. GEMM: 64x128 block tile, each wave a 32x32 micro-tile
// (2 A-frags x 2 B-frags -> 4 WMMAs per K-step) for WMMA density and halved
// L2 weight traffic. Fragment layouts per CDNA5 ISA.
// ---------------------------------------------------------------------------

typedef __attribute__((ext_vector_type(16))) __bf16 v16bf;
typedef __attribute__((ext_vector_type(8)))  __bf16 v8bf;
typedef __attribute__((ext_vector_type(8)))  float  v8f;

#define WMMA_BF16(a, b, c) \
  __builtin_amdgcn_wmma_f32_16x16x32_bf16(false, (a), false, (b), (short)0, (c), false, false)

// ------------------------------- f32 -> bf16 -------------------------------
__global__ __launch_bounds__(256) void cvt_bf16_kernel(const float* __restrict__ src,
                                                       __bf16* __restrict__ dst,
                                                       size_t n) {
  size_t stride = (size_t)gridDim.x * blockDim.x;
  for (size_t i = (size_t)blockIdx.x * blockDim.x + threadIdx.x; i < n; i += stride)
    dst[i] = (__bf16)src[i];
}

// --------------------------- out = res + LN(y) -----------------------------
__global__ __launch_bounds__(256) void add_ln_kernel(const float* __restrict__ res,
                                                     const float* __restrict__ y,
                                                     const float* __restrict__ gam,
                                                     const float* __restrict__ bet,
                                                     float* __restrict__ of32,
                                                     __bf16* __restrict__ obf) {
  const int H = 4096;
  const int row = blockIdx.x;
  const float* yr = y + (size_t)row * H;
  float s = 0.f, s2 = 0.f;
  for (int i = threadIdx.x; i < H; i += 256) { float v = yr[i]; s += v; s2 += v * v; }
#pragma unroll
  for (int m = 16; m >= 1; m >>= 1) { s += __shfl_xor(s, m, 32); s2 += __shfl_xor(s2, m, 32); }
  __shared__ float rs[8], rq[8];
  if ((threadIdx.x & 31) == 0) { rs[threadIdx.x >> 5] = s; rq[threadIdx.x >> 5] = s2; }
  __syncthreads();
  s = 0.f; s2 = 0.f;
#pragma unroll
  for (int i = 0; i < 8; ++i) { s += rs[i]; s2 += rq[i]; }
  const float mu = s * (1.0f / H);
  const float var = s2 * (1.0f / H) - mu * mu;
  const float rinv = rsqrtf(var + 1e-5f);
  for (int i = threadIdx.x; i < H; i += 256) {
    float v = (yr[i] - mu) * rinv * gam[i] + bet[i];
    if (res) v += res[(size_t)row * H + i];
    if (of32) of32[(size_t)row * H + i] = v;
    obf[(size_t)row * H + i] = (__bf16)v;
  }
}

// ------------------------------ WMMA GEMM ----------------------------------
// C[M,N] = A[M,K] bf16 @ W[K,N] bf16 + bias, optional exact GELU; f32 and/or
// bf16 out. Block: 256 thr = 8 waves (2x4). Block tile 64x128, wave tile
// 32x32 (2x2 fragments), K-step 32.
template <int GELU, int OF32, int OBF16>
__global__ __launch_bounds__(256) void gemm_wmma_kernel(const __bf16* __restrict__ A,
                                                        const __bf16* __restrict__ W,
                                                        const float* __restrict__ bias,
                                                        float* __restrict__ Cf,
                                                        __bf16* __restrict__ Cb,
                                                        int M, int N, int K) {
  constexpr int BM = 64, BN = 128, BK = 32;
  __shared__ __bf16 sA[BM][BK + 8];
  __shared__ __bf16 sB[BK][BN + 8];
  const int tid = threadIdx.x;
  const int lane = tid & 31, wave = tid >> 5;
  const int wm = wave >> 2, wn = wave & 3;
  const int lm = lane & 15, hf = lane >> 4;
  const int mbase = blockIdx.y * BM, nbase = blockIdx.x * BN;

  v8f acc00 = {0.f, 0.f, 0.f, 0.f, 0.f, 0.f, 0.f, 0.f};
  v8f acc01 = acc00, acc10 = acc00, acc11 = acc00;

  for (int k0 = 0; k0 < K; k0 += BK) {
    {  // A tile: 64x32 = 2048 elems, 8 per thread (16B vectors)
      int idx = tid * 8;
      int r = idx >> 5, c = idx & 31;
      *(v8bf*)&sA[r][c] = *(const v8bf*)&A[(size_t)(mbase + r) * K + k0 + c];
    }
#pragma unroll
    for (int p = 0; p < 2; ++p) {  // B tile: 32x128 = 4096 elems, 16B vectors
      int idx = (tid + p * 256) * 8;
      int r = idx >> 7, c = idx & 127;
      *(v8bf*)&sB[r][c] = *(const v8bf*)&W[(size_t)(k0 + r) * N + nbase + c];
    }
    if (k0 + BK < K)  // CDNA5 global_prefetch_b8 for next W tile
      __builtin_prefetch(&W[(size_t)(k0 + BK + (tid >> 3)) * N + nbase + (tid & 7) * 16], 0, 0);
    __syncthreads();

    // A fragments 16x32 bf16: lane half picks K base 0/8; elems 8..15 are K+16.
    v16bf af0, af1;
    {
      const int kb = hf * 8;
#pragma unroll
      for (int i = 0; i < 8; ++i) {
        af0[i]     = sA[wm * 32 + lm][kb + i];
        af0[8 + i] = sA[wm * 32 + lm][16 + kb + i];
        af1[i]     = sA[wm * 32 + 16 + lm][kb + i];
        af1[8 + i] = sA[wm * 32 + 16 + lm][16 + kb + i];
      }
    }
    // B fragments 32x16: lane = column, half picks K base 0/8; elems 8..15 are K+16.
    v16bf b0, b1;
#pragma unroll
    for (int i = 0; i < 16; ++i) {
      b0[i] = sB[hf * 8 + (i & 7) + (i >> 3) * 16][wn * 32 + lm];
      b1[i] = sB[hf * 8 + (i & 7) + (i >> 3) * 16][wn * 32 + 16 + lm];
    }
    acc00 = WMMA_BF16(af0, b0, acc00);
    acc01 = WMMA_BF16(af0, b1, acc01);
    acc10 = WMMA_BF16(af1, b0, acc10);
    acc11 = WMMA_BF16(af1, b1, acc11);
    __syncthreads();
  }

  const int c0 = nbase + wn * 32 + lm;
  const int c1 = c0 + 16;
  const float bia0 = bias[c0], bia1 = bias[c1];
#pragma unroll
  for (int mt = 0; mt < 2; ++mt) {
    v8f a0 = mt ? acc10 : acc00;
    v8f a1 = mt ? acc11 : acc01;
#pragma unroll
    for (int r = 0; r < 8; ++r) {
      const int row = mbase + wm * 32 + mt * 16 + r + 8 * hf;  // C rows r, r+8
      float v0 = a0[r] + bia0;
      float v1 = a1[r] + bia1;
      if (GELU) {
        v0 = 0.5f * v0 * (1.0f + erff(v0 * 0.70710678f));
        v1 = 0.5f * v1 * (1.0f + erff(v1 * 0.70710678f));
      }
      size_t o0 = (size_t)row * N + c0, o1 = (size_t)row * N + c1;
      if (OF32) { Cf[o0] = v0; Cf[o1] = v1; }
      if (OBF16) { Cb[o0] = (__bf16)v0; Cb[o1] = (__bf16)v1; }
    }
  }
}

// --------------------------- flash GQA attention ---------------------------
// One wave32 per (b, g, h, 16-row q tile). Q [B,S,H] bf16 head-packed,
// K/V [B,S,G*D] bf16. Online softmax across causal K chunks of 32.
__global__ __launch_bounds__(32) void attn_flash_kernel(const __bf16* __restrict__ Q,
                                                        const __bf16* __restrict__ Kp,
                                                        const __bf16* __restrict__ Vp,
                                                        __bf16* __restrict__ O) {
  const int S = 1024, H = 4096, D = 128, HG = 16, GD = 256;
  const int tile = blockIdx.x;
  const int qt = tile & 63;
  const int h  = (tile >> 6) & 15;
  const int g  = (tile >> 10) & 1;
  const int b  = tile >> 11;
  const int lane = threadIdx.x;
  const int lm = lane & 15, hf = lane >> 4;
  const int q_base = qt * 16;
  const float scale = 0.088388347648318447f;  // 1/sqrt(128)

  // Q A-fragments for all 4 d-chunks (contracting D=128 in steps of 32)
  const __bf16* qrow = Q + ((size_t)(b * S + q_base + lm)) * H + (g * HG + h) * D;
  v16bf qf[4];
#pragma unroll
  for (int c = 0; c < 4; ++c) {
    const int d0 = c * 32 + hf * 16;
#pragma unroll
    for (int i = 0; i < 8; ++i) { qf[c][i] = qrow[d0 + i]; qf[c][8 + i] = qrow[d0 + 8 + i]; }
  }

  v8f zero8 = {0.f, 0.f, 0.f, 0.f, 0.f, 0.f, 0.f, 0.f};
  v8f oacc[8];
#pragma unroll
  for (int dt = 0; dt < 8; ++dt) oacc[dt] = zero8;
  float mrow[8], lrow[8];
#pragma unroll
  for (int r = 0; r < 8; ++r) { mrow[r] = -1e30f; lrow[r] = 0.f; }

  __shared__ __bf16 sp[16][32];  // probability tile for lane-transpose

  for (int c = 0; c * 32 <= q_base; ++c) {
    const int kbase = c * 32;
    // scores: Q (16x128) x K^T chunk (128x32) as two 16x16 tiles
    v8f s0 = zero8, s1 = zero8;
#pragma unroll
    for (int dc = 0; dc < 4; ++dc) {
      v16bf b0, b1;
      const __bf16* k0 = Kp + ((size_t)(b * S + kbase + lm)) * GD + g * D + dc * 32 + hf * 16;
      const __bf16* k1 = Kp + ((size_t)(b * S + kbase + 16 + lm)) * GD + g * D + dc * 32 + hf * 16;
#pragma unroll
      for (int i = 0; i < 16; ++i) { b0[i] = k0[i]; b1[i] = k1[i]; }
      s0 = WMMA_BF16(qf[dc], b0, s0);
      s1 = WMMA_BF16(qf[dc], b1, s1);
    }
    // causal mask + online softmax (per-row stats across each 16-lane half)
    float alpha[8];
#pragma unroll
    for (int r = 0; r < 8; ++r) {
      const int row = q_base + r + 8 * hf;
      float x0 = s0[r] * scale; if (kbase + lm > row) x0 = -1e30f;
      float x1 = s1[r] * scale; if (kbase + 16 + lm > row) x1 = -1e30f;
      float mx = fmaxf(x0, x1);
#pragma unroll
      for (int mb = 1; mb <= 8; mb <<= 1) mx = fmaxf(mx, __shfl_xor(mx, mb, 32));
      const float mnew = fmaxf(mrow[r], mx);
      const float a = __expf(mrow[r] - mnew);
      const float p0 = __expf(x0 - mnew);
      const float p1 = __expf(x1 - mnew);
      float ps = p0 + p1;
#pragma unroll
      for (int mb = 1; mb <= 8; mb <<= 1) ps += __shfl_xor(ps, mb, 32);
      lrow[r] = lrow[r] * a + ps;
      mrow[r] = mnew;
      alpha[r] = a;
      sp[r + 8 * hf][lm]      = (__bf16)p0;
      sp[r + 8 * hf][16 + lm] = (__bf16)p1;
    }
#pragma unroll
    for (int dt = 0; dt < 8; ++dt)
#pragma unroll
      for (int r = 0; r < 8; ++r) oacc[dt][r] *= alpha[r];
    __syncthreads();
    // P as A-fragment (16 q rows x 32 k), via LDS transpose
    v16bf pa;
    {
      const int kb = hf * 16;
#pragma unroll
      for (int i = 0; i < 8; ++i) { pa[i] = sp[lm][kb + i]; pa[8 + i] = sp[lm][kb + 8 + i]; }
    }
    __syncthreads();
    // out += P (16x32) x V chunk (32x128) across 8 d-tiles
#pragma unroll
    for (int dt = 0; dt < 8; ++dt) {
      v16bf bv;
      const __bf16* vp = Vp + ((size_t)(b * S + kbase + hf * 16)) * GD + g * D + dt * 16 + lm;
#pragma unroll
      for (int i = 0; i < 16; ++i) bv[i] = vp[(size_t)i * GD];
      oacc[dt] = WMMA_BF16(pa, bv, oacc[dt]);
    }
  }

#pragma unroll
  for (int r = 0; r < 8; ++r) {
    const float inv = 1.0f / lrow[r];
    const int row = q_base + r + 8 * hf;
#pragma unroll
    for (int dt = 0; dt < 8; ++dt)
      O[((size_t)(b * S + row)) * H + (g * HG + h) * D + dt * 16 + lm] =
          (__bf16)(oacc[dt][r] * inv);
  }
}

// ------------------------------- launcher ----------------------------------
extern "C" void kernel_launch(void* const* d_in, const int* in_sizes, int n_in,
                              void* d_out, int out_size, void* d_ws, size_t ws_size,
                              hipStream_t stream) {
  (void)in_sizes; (void)n_in; (void)out_size; (void)ws_size;
  const float* x    = (const float*)d_in[0];
  const float* ln1g = (const float*)d_in[1];
  const float* ln1b = (const float*)d_in[2];
  const float* wq   = (const float*)d_in[3];
  const float* bq   = (const float*)d_in[4];
  const float* wk   = (const float*)d_in[5];
  const float* bk   = (const float*)d_in[6];
  const float* wv   = (const float*)d_in[7];
  const float* bv   = (const float*)d_in[8];
  const float* wo   = (const float*)d_in[9];
  const float* bo   = (const float*)d_in[10];
  const float* ln2g = (const float*)d_in[11];
  const float* ln2b = (const float*)d_in[12];
  const float* wup  = (const float*)d_in[13];
  const float* bup  = (const float*)d_in[14];
  const float* wdn  = (const float*)d_in[15];
  const float* bdn  = (const float*)d_in[16];

  const int M = 2048, H = 4096, GDm = 256, F = 16384;

  char* ws = (char*)d_ws;
  size_t off = 0;
  auto alloc = [&](size_t bytes) -> void* {
    void* p = ws + off;
    off = (off + bytes + 255) & ~(size_t)255;
    return p;
  };
  float*  x1f  = (float*) alloc((size_t)M * H * 4);
  __bf16* x1b  = (__bf16*)alloc((size_t)M * H * 2);
  __bf16* qb   = (__bf16*)alloc((size_t)M * H * 2);
  __bf16* kbuf = (__bf16*)alloc((size_t)M * GDm * 2);
  __bf16* vbuf = (__bf16*)alloc((size_t)M * GDm * 2);
  __bf16* ab   = (__bf16*)alloc((size_t)M * H * 2);
  float*  apf  = (float*) alloc((size_t)M * H * 4);
  __bf16* x2b  = (__bf16*)alloc((size_t)M * H * 2);
  __bf16* hb   = (__bf16*)alloc((size_t)M * F * 2);
  __bf16* wqb  = (__bf16*)alloc((size_t)H * H * 2);
  __bf16* wkb  = (__bf16*)alloc((size_t)H * GDm * 2);
  __bf16* wvb  = (__bf16*)alloc((size_t)H * GDm * 2);
  __bf16* wob  = (__bf16*)alloc((size_t)H * H * 2);
  __bf16* wupb = (__bf16*)alloc((size_t)H * F * 2);
  __bf16* wdnb = (__bf16*)alloc((size_t)F * H * 2);

  // weights -> bf16 (so w_up/w_dn fit L2 across M-tile re-reads)
  cvt_bf16_kernel<<<8192, 256, 0, stream>>>(wq,  wqb,  (size_t)H * H);
  cvt_bf16_kernel<<<2048, 256, 0, stream>>>(wk,  wkb,  (size_t)H * GDm);
  cvt_bf16_kernel<<<2048, 256, 0, stream>>>(wv,  wvb,  (size_t)H * GDm);
  cvt_bf16_kernel<<<8192, 256, 0, stream>>>(wo,  wob,  (size_t)H * H);
  cvt_bf16_kernel<<<8192, 256, 0, stream>>>(wup, wupb, (size_t)H * F);
  cvt_bf16_kernel<<<8192, 256, 0, stream>>>(wdn, wdnb, (size_t)F * H);

  // x1 = x + LN1(x)
  add_ln_kernel<<<M, 256, 0, stream>>>(x, x, ln1g, ln1b, x1f, x1b);

  dim3 gHH(H / 128, M / 64);
  dim3 gKV(GDm / 128, M / 64);
  dim3 gUP(F / 128, M / 64);

  gemm_wmma_kernel<0, 0, 1><<<gHH, 256, 0, stream>>>(x1b, wqb, bq, nullptr, qb,   M, H,   H);
  gemm_wmma_kernel<0, 0, 1><<<gKV, 256, 0, stream>>>(x1b, wkb, bk, nullptr, kbuf, M, GDm, H);
  gemm_wmma_kernel<0, 0, 1><<<gKV, 256, 0, stream>>>(x1b, wvb, bv, nullptr, vbuf, M, GDm, H);

  attn_flash_kernel<<<4096, 32, 0, stream>>>(qb, kbuf, vbuf, ab);

  gemm_wmma_kernel<0, 1, 0><<<gHH, 256, 0, stream>>>(ab, wob, bo, apf, nullptr, M, H, H);

  // x2 = x1 + LN2(attn_out)
  add_ln_kernel<<<M, 256, 0, stream>>>(x1f, apf, ln2g, ln2b, nullptr, x2b);

  gemm_wmma_kernel<1, 0, 1><<<gUP, 256, 0, stream>>>(x2b, wupb, bup, nullptr, hb, M, F, H);
  gemm_wmma_kernel<0, 1, 0><<<gHH, 256, 0, stream>>>(hb, wdnb, bdn, (float*)d_out, nullptr, M, H, F);
}
